// CantorMoELayer_58677843198310
// MI455X (gfx1250) — matrix operand order repaired and checked by reading, not compiled, not run on hardware
//
#include <hip/hip_runtime.h>
#include <hip/hip_bf16.h>
#include <math.h>

// ---------------------------------------------------------------------------
// CantorMoE layer for MI455X (gfx1250, wave32, WMMA).
// B=8 P=2048 D=1024 E=16 S=64 DK=128 H=16. HBM-bound (~128MB traffic);
// GEMMs (f@wv, V@wout) run on v_wmma_f32_16x16x32_f16 with weights pre-packed
// into per-lane WMMA B-fragment layout in d_ws (L2 resident, 192MB L2).
// ---------------------------------------------------------------------------

typedef _Float16 v16h __attribute__((ext_vector_type(16)));
typedef _Float16 h2   __attribute__((ext_vector_type(2)));
typedef float    v8f  __attribute__((ext_vector_type(8)));

#define E_  16
#define S_  64
#define DK_ 128
#define H_  16
#define D_  1024
#define P_  2048
#define FSTR 1032   // feats LDS row stride in halves (pad: spreads banks)
#define SSTR 72     // stage LDS row stride in halves

// ws layout (bytes):
//   [0      , 262144) : wv  packed f16 B-fragments  [E][2ks][8nt][32lane][8dw]
//   [262144 , 524288) : wout packed f16 B-fragments [E][4ks][4nt][32lane][8dw]
//   [524288 , 532480) : dsum[E][DK] f32  (sum of normalized penta dirs)
//   [532480 , 532544) : w_e[E] f32
//   [532544 , 532608) : aw[E]  f32 (sigmoid(alpha))
#define WS_B2   262144
#define WS_DSUM 524288
#define WS_WE   532480
#define WS_AW   532544

union FragU {
  uint4     q[2];
  h2        h[8];
  v16h      v;
  _Float16  f[16];
};

__device__ __forceinline__ float sigm_(float x) { return 1.0f / (1.0f + __expf(-x)); }

#define LDS_FENCE() asm volatile("s_wait_dscnt 0" ::: "memory")

// ---------------- prep: per-expert scalars + zero dsum ----------------------
__global__ void prep_scalars(const float* __restrict__ alpha,
                             const float* __restrict__ betas,
                             const float* __restrict__ pos_embed,
                             float* __restrict__ dsum,
                             float* __restrict__ wE,
                             float* __restrict__ wAW) {
  int t = threadIdx.x;
  for (int i = t; i < E_ * DK_; i += 256) dsum[i] = 0.0f;
  if (t < E_) {
    float s = 0.0f;
    for (int i = 0; i < DK_; ++i) s += pos_embed[t * DK_ + i];
    float pos_w = sigm_(s * (1.0f / DK_));
    const int offs[4] = {-2, -1, 1, 2};
    float bs = 0.0f, vc = 0.0f;
    for (int j = 0; j < 4; ++j) {
      int nb = t + offs[j];
      if (nb >= 0 && nb < E_) { bs += sigm_(betas[t * 4 + j]); vc += 1.0f; }
    }
    wE[t]  = pos_w * (1.0f + bs / vc);
    wAW[t] = sigm_(alpha[t]);
  }
}

// ---------------- prep: dsum[e][d] = sum_v penta_normalized -----------------
__global__ void prep_dirs(const float* __restrict__ penta, float* __restrict__ dsum) {
  int r = blockIdx.x;            // r = e*5 + v, 80 rows
  int e = r / 5;
  int lane = threadIdx.x;        // 32 lanes
  float v0 = penta[r * DK_ + lane +  0];
  float v1 = penta[r * DK_ + lane + 32];
  float v2 = penta[r * DK_ + lane + 64];
  float v3 = penta[r * DK_ + lane + 96];
  float ss = v0 * v0 + v1 * v1 + v2 * v2 + v3 * v3;
  ss += __shfl_xor(ss, 16, 32);
  ss += __shfl_xor(ss, 8, 32);
  ss += __shfl_xor(ss, 4, 32);
  ss += __shfl_xor(ss, 2, 32);
  ss += __shfl_xor(ss, 1, 32);
  float rn = rsqrtf(ss);
  atomicAdd(&dsum[e * DK_ + lane +  0], v0 * rn);
  atomicAdd(&dsum[e * DK_ + lane + 32], v1 * rn);
  atomicAdd(&dsum[e * DK_ + lane + 64], v2 * rn);
  atomicAdd(&dsum[e * DK_ + lane + 96], v3 * rn);
}

// ---------------- prep: pack wv/wout into WMMA B-fragment layout ------------
// B 32x16 f16 layout (wave32): lanes 0-15 hold N=lane, K=0..7,16..23 (VGPR j = K{2j,2j+1}, +8 for j>=4);
// lanes 16-31 hold N=lane-16, K=8..15,24..31.
__global__ void prep_pack(const float* __restrict__ wv,
                          const float* __restrict__ wout,
                          unsigned* __restrict__ wsB1,
                          unsigned* __restrict__ wsB2) {
  int idx = blockIdx.x * 256 + threadIdx.x;   // 512*256 = 131072 total
  union { unsigned u; _Float16 f[2]; } Pk;
  if (idx < 65536) {
    int j = idx & 7, lane = (idx >> 3) & 31, nt = (idx >> 8) & 7;
    int ks = (idx >> 11) & 1, e = idx >> 12;
    int k = ks * 32 + ((lane >> 4) << 3) + 2 * j + ((j >> 2) << 3);
    int n = nt * 16 + (lane & 15);
    Pk.f[0] = (_Float16)wv[(e * S_ + k)     * DK_ + n];
    Pk.f[1] = (_Float16)wv[(e * S_ + k + 1) * DK_ + n];
    wsB1[idx] = Pk.u;
  } else {
    int i2 = idx - 65536;
    int j = i2 & 7, lane = (i2 >> 3) & 31, nt = (i2 >> 8) & 3;
    int ks = (i2 >> 10) & 3, e = i2 >> 12;
    int k = ks * 32 + ((lane >> 4) << 3) + 2 * j + ((j >> 2) << 3);
    int n = nt * 16 + (lane & 15);
    Pk.f[0] = (_Float16)wout[(e * DK_ + k)     * S_ + n];
    Pk.f[1] = (_Float16)wout[(e * DK_ + k + 1) * S_ + n];
    wsB2[i2] = Pk.u;
  }
}

// ---------------- main kernel: 16 tokens / block, 2 experts / wave ----------
__launch_bounds__(256, 1)
__global__ void cantor_moe_main(const float* __restrict__ x,
                                const float* __restrict__ fingerprints,
                                const float* __restrict__ gamma,
                                const float* __restrict__ lbeta,
                                const float* __restrict__ w1,
                                const float* __restrict__ b1,
                                const float* __restrict__ w2,
                                const float* __restrict__ b2,
                                const float* __restrict__ temp,
                                const unsigned* __restrict__ wsB1,
                                const unsigned* __restrict__ wsB2,
                                const float* __restrict__ dsum,
                                const float* __restrict__ wE,
                                const float* __restrict__ wAW,
                                float* __restrict__ out) {
  __shared__ __align__(16) _Float16 feats[16 * FSTR];     // normalized x, f16
  __shared__ __align__(16) _Float16 stage[8 * 16 * SSTR]; // per-wave transpose stage
  __shared__ float redA[256], redB[256];
  __shared__ float numL[16];
  __shared__ float invden[16];
  __shared__ unsigned mbits[16];

  const int t   = threadIdx.x;
  const int m0  = t >> 4, sub = t & 15;
  const long tokBase = (long)blockIdx.x * 16;

  // ---- Phase 0: LayerNorm (two-pass over L2-hot x), f16 feats to LDS ----
  {
    const float4* xr = (const float4*)(x + (tokBase + m0) * D_ + sub * 64);
    float s = 0.0f, s2 = 0.0f;
#pragma unroll
    for (int i = 0; i < 16; ++i) {
      float4 v = xr[i];
      s  += v.x + v.y + v.z + v.w;
      s2 += v.x * v.x + v.y * v.y + v.z * v.z + v.w * v.w;
    }
    redA[t] = s; redB[t] = s2;
    __syncthreads();
    float mu = 0.0f, vv = 0.0f;
#pragma unroll
    for (int k = 0; k < 16; ++k) { mu += redA[m0 * 16 + k]; vv += redB[m0 * 16 + k]; }
    mu *= (1.0f / 1024.0f);
    vv = vv * (1.0f / 1024.0f) - mu * mu;
    float rs = rsqrtf(vv + 1e-5f);
#pragma unroll
    for (int i = 0; i < 16; ++i) {
      float4 v = xr[i];
      int db = sub * 64 + i * 4;
      feats[m0 * FSTR + db + 0] = (_Float16)((v.x - mu) * rs * gamma[db + 0] + lbeta[db + 0]);
      feats[m0 * FSTR + db + 1] = (_Float16)((v.y - mu) * rs * gamma[db + 1] + lbeta[db + 1]);
      feats[m0 * FSTR + db + 2] = (_Float16)((v.z - mu) * rs * gamma[db + 2] + lbeta[db + 2]);
      feats[m0 * FSTR + db + 3] = (_Float16)((v.w - mu) * rs * gamma[db + 3] + lbeta[db + 3]);
    }
    if (t < 16) {                       // fingerprint bands + fusion denominator
      int p = (int)((tokBase + t) % P_);
      float fp = fingerprints[p];
      float wsum = 0.0f; unsigned bits = 0u;
      for (int e = 0; e < E_; ++e) {
        float fmn = fmaxf(0.0f, e * (1.0f / E_) - 0.03125f);
        float fmx = fminf(1.0f, (e + 1) * (1.0f / E_) + 0.03125f);
        if (fp >= fmn && fp < fmx) { wsum += wE[e]; bits |= (1u << e); }
      }
      mbits[t] = bits;
      invden[t] = 1.0f / fmaxf(wsum, 1e-6f);
      numL[t] = 0.0f;
    }
  }
  __syncthreads();

  const int wave = t >> 5, lane = t & 31;
  const int lo = lane & 15, hi = lane >> 4;
  const int e0 = wave * 2;

  __builtin_prefetch(wsB1 + (e0 * 2) * 8 * 256, 0, 0);  // global_prefetch_b8
  __builtin_prefetch(wsB2 + (e0 * 4) * 4 * 256, 0, 0);

  // ---- alpha gate MLP: lane -> (token=lo, expert=e0+hi), VALU ----
  float gscale;
  {
    const int ge = e0 + hi;
    float acc[H_];
#pragma unroll
    for (int h = 0; h < H_; ++h) acc[h] = b1[ge * H_ + h];
    for (int s = 0; s < S_; ++s) {
      float f = (float)feats[lo * FSTR + ge * 64 + s];
#pragma unroll
      for (int h = 0; h < H_; ++h) acc[h] = fmaf(f, w1[(ge * S_ + s) * H_ + h], acc[h]);
    }
    float s2 = b2[ge];
#pragma unroll
    for (int h = 0; h < H_; ++h) {
      float a = acc[h];
      float g = 0.5f * a * (1.0f + erff(a * 0.70710678118654752f));   // exact GELU
      s2 = fmaf(g, w2[ge * H_ + h], s2);
    }
    float awv = wAW[ge];
    gscale = sigm_(s2) * awv + (1.0f - awv);
  }

  // ---- V = (feats * gscale) @ wv : WMMA f16 -> f32, C kept in registers ----
  v8f Vc[2][8];
#pragma unroll
  for (int ei = 0; ei < 2; ++ei)
#pragma unroll
    for (int nt = 0; nt < 8; ++nt) {
      v8f z = {0.0f, 0.0f, 0.0f, 0.0f, 0.0f, 0.0f, 0.0f, 0.0f};
      Vc[ei][nt] = z;
    }

#pragma unroll
  for (int ei = 0; ei < 2; ++ei) {
    const int e = e0 + ei;
    float gme = __shfl(gscale, (ei << 4) | lo, 32);   // gate for my A-row
    h2 gh; gh[0] = (_Float16)gme; gh[1] = (_Float16)gme;
#pragma unroll
    for (int ks = 0; ks < 2; ++ks) {
      FragU A;   // A 16x32 f16: lane<16 -> M=lane,K{0..7,16..23}; lane>=16 -> K{8..15,24..31}
      const char* fb = (const char*)feats + lo * (FSTR * 2) + (e * 64 + ks * 32) * 2 + (hi ? 16 : 0);
      A.q[0] = *(const uint4*)fb;
      A.q[1] = *(const uint4*)(fb + 32);
#pragma unroll
      for (int k2 = 0; k2 < 8; ++k2) A.h[k2] = A.h[k2] * gh;   // v_pk_mul_f16
#pragma unroll
      for (int nt = 0; nt < 8; ++nt) {
        FragU Bf;
        const uint4* bp = (const uint4*)((const char*)wsB1 + ((((e * 2 + ks) * 8 + nt) * 32 + lane) * 32));
        Bf.q[0] = bp[0]; Bf.q[1] = bp[1];
        Vc[ei][nt] = __builtin_amdgcn_wmma_f32_16x16x32_f16(
            false, A.v, false, Bf.v, (short)0, Vc[ei][nt], false, false);
      }
    }
  }

  // ---- q_e = V . dsum_e per token; masked weighted sum into numL ----
#pragma unroll
  for (int ei = 0; ei < 2; ++ei) {
    const int e = e0 + ei;
    float qp[8];
#pragma unroll
    for (int r = 0; r < 8; ++r) qp[r] = 0.0f;
#pragma unroll
    for (int nt = 0; nt < 8; ++nt) {
      float dv = dsum[e * DK_ + nt * 16 + lo];
#pragma unroll
      for (int r = 0; r < 8; ++r) qp[r] = fmaf(Vc[ei][nt][r], dv, qp[r]);
    }
#pragma unroll
    for (int r = 0; r < 8; ++r) {
      qp[r] += __shfl_xor(qp[r], 1, 16);
      qp[r] += __shfl_xor(qp[r], 2, 16);
      qp[r] += __shfl_xor(qp[r], 4, 16);
      qp[r] += __shfl_xor(qp[r], 8, 16);
    }
    if (lo < 8) {
      int mt = hi * 8 + lo;
      float qv = qp[0];
      if (lo == 1) qv = qp[1];
      if (lo == 2) qv = qp[2];
      if (lo == 3) qv = qp[3];
      if (lo == 4) qv = qp[4];
      if (lo == 5) qv = qp[5];
      if (lo == 6) qv = qp[6];
      if (lo == 7) qv = qp[7];
      if ((mbits[mt] >> e) & 1u) atomicAdd(&numL[mt], qv * wE[e]);  // ds_add_f32
    }
  }
  __syncthreads();

  // ---- fused scalar per token ----
  const float invT = 1.0f / fabsf(temp[0]);
  float fr[8];
#pragma unroll
  for (int r = 0; r < 8; ++r) {
    int mt = hi * 8 + r;
    fr[r] = numL[mt] * 0.2f * invden[mt] * invT;
  }

  // ---- rec = (V*fused) @ wout : C-layout -> A-layout via LDS, then WMMA ----
  _Float16* stw = stage + wave * (16 * SSTR);
#pragma unroll
  for (int ei = 0; ei < 2; ++ei) {
    const int e = e0 + ei;
    v8f rc[4];
#pragma unroll
    for (int nt = 0; nt < 4; ++nt) {
      v8f z = {0.0f, 0.0f, 0.0f, 0.0f, 0.0f, 0.0f, 0.0f, 0.0f};
      rc[nt] = z;
    }
#pragma unroll
    for (int dh = 0; dh < 2; ++dh) {          // two 64-wide K halves of DK=128
      LDS_FENCE();                             // stage reads of prev iter done
#pragma unroll
      for (int nt = 0; nt < 4; ++nt) {
        int gnt = dh * 4 + nt;
#pragma unroll
        for (int r = 0; r < 8; ++r) {
          int mt = hi * 8 + r;
          stw[mt * SSTR + nt * 16 + lo] = (_Float16)(Vc[ei][gnt][r] * fr[r]);
        }
      }
      LDS_FENCE();                             // stage stores visible wave-wide
#pragma unroll
      for (int ksl = 0; ksl < 2; ++ksl) {
        int ks = dh * 2 + ksl;
        FragU A;
        const char* sb = (const char*)stw + lo * (SSTR * 2) + (ksl * 32 + (hi ? 8 : 0)) * 2;
        A.q[0] = *(const uint4*)sb;
        A.q[1] = *(const uint4*)(sb + 32);
#pragma unroll
        for (int nt = 0; nt < 4; ++nt) {
          FragU Bf;
          const uint4* bp = (const uint4*)((const char*)wsB2 + ((((e * 4 + ks) * 4 + nt) * 32 + lane) * 32));
          Bf.q[0] = bp[0]; Bf.q[1] = bp[1];
          rc[nt] = __builtin_amdgcn_wmma_f32_16x16x32_f16(
              false, A.v, false, Bf.v, (short)0, rc[nt], false, false);
        }
      }
    }
    // ---- masked residual scatter: out = x + rec * mask ----
#pragma unroll
    for (int nt = 0; nt < 4; ++nt) {
#pragma unroll
      for (int r = 0; r < 8; ++r) {
        int mt = hi * 8 + r;
        float msk = (float)((mbits[mt] >> e) & 1u);
        long gi = (tokBase + mt) * D_ + e * 64 + nt * 16 + lo;
        out[gi] = x[gi] + rc[nt][r] * msk;
      }
    }
  }
}

// ---------------------------------------------------------------------------
extern "C" void kernel_launch(void* const* d_in, const int* in_sizes, int n_in,
                              void* d_out, int out_size, void* d_ws, size_t ws_size,
                              hipStream_t stream) {
  (void)in_sizes; (void)n_in; (void)out_size; (void)ws_size;  // needs >= 532608 B ws
  const float* x     = (const float*)d_in[0];
  const float* fprt  = (const float*)d_in[1];
  const float* gam   = (const float*)d_in[2];
  const float* bet   = (const float*)d_in[3];
  const float* w1    = (const float*)d_in[4];
  const float* b1    = (const float*)d_in[5];
  const float* w2    = (const float*)d_in[6];
  const float* b2    = (const float*)d_in[7];
  const float* alpha = (const float*)d_in[8];
  const float* wv    = (const float*)d_in[9];
  const float* penta = (const float*)d_in[10];
  const float* betas = (const float*)d_in[11];
  const float* wout  = (const float*)d_in[12];
  const float* pose  = (const float*)d_in[13];
  const float* temp  = (const float*)d_in[14];

  char* ws = (char*)d_ws;
  unsigned* wsB1 = (unsigned*)ws;
  unsigned* wsB2 = (unsigned*)(ws + WS_B2);
  float* dsum = (float*)(ws + WS_DSUM);
  float* wEp  = (float*)(ws + WS_WE);
  float* wAWp = (float*)(ws + WS_AW);

  prep_scalars<<<1, 256, 0, stream>>>(alpha, betas, pose, dsum, wEp, wAWp);
  prep_dirs<<<80, 32, 0, stream>>>(penta, dsum);
  prep_pack<<<512, 256, 0, stream>>>(wv, wout, wsB1, wsB2);
  cantor_moe_main<<<1024, 256, 0, stream>>>(x, fprt, gam, bet, w1, b1, w2, b2,
                                            temp, wsB1, wsB2, dsum, wEp, wAWp,
                                            (float*)d_out);
}
